// CrossAttentionFusion_23278722745037
// MI455X (gfx1250) — compile-verified, matrix operand order checked by reading, not run on hardware
//
#include <hip/hip_runtime.h>

// ---------------------------------------------------------------------------
// CDNA5 (gfx1250) cross-attention fusion.
// All matmuls via v_wmma_f32_16x16x32_bf16 (wave32, 16x16 f32 C/D = 8 VGPRs).
// ---------------------------------------------------------------------------

typedef __bf16 v16bf __attribute__((ext_vector_type(16)));
typedef __bf16 v8bf  __attribute__((ext_vector_type(8)));
typedef float  v8f   __attribute__((ext_vector_type(8)));

union F8   { v8f v; float f[8]; };
union BF16F { v16bf v; __bf16 e[16]; v8bf h[2]; };
union BF8P  { v8bf v; __bf16 e[8]; };

__device__ __forceinline__ __bf16 f2bf(float x) {
    unsigned u = __builtin_bit_cast(unsigned, x);
    unsigned r = (u + 0x7FFFu + ((u >> 16) & 1u)) >> 16;   // round-to-nearest-even
    unsigned short s = (unsigned short)r;
    return __builtin_bit_cast(__bf16, s);
}

__device__ __forceinline__ v8f zero8() {
    F8 z;
#pragma unroll
    for (int i = 0; i < 8; ++i) z.f[i] = 0.0f;
    return z.v;
}

// Fragment = 16 bf16 per lane: elems 0-7 <- p0[0..7], elems 8-15 <- p1[0..7].
__device__ __forceinline__ v16bf load_bf16_frag(const __bf16* p0, const __bf16* p1) {
    BF16F r;
    r.h[0] = *reinterpret_cast<const v8bf*>(p0);
    r.h[1] = *reinterpret_cast<const v8bf*>(p1);
    return r.v;
}

__device__ __forceinline__ v16bf load_f32_frag(const float* p0, const float* p1) {
    BF16F r;
#pragma unroll
    for (int i = 0; i < 8; ++i) r.e[i]     = f2bf(p0[i]);
#pragma unroll
    for (int i = 0; i < 8; ++i) r.e[8 + i] = f2bf(p1[i]);
    return r.v;
}

__device__ __forceinline__ v8f wmma_bf16(v16bf a, v16bf b, v8f c) {
    // (neg_a, A, neg_b, B, c_mod, C, reuse_a, reuse_b)
    return __builtin_amdgcn_wmma_f32_16x16x32_bf16(false, a, false, b, (short)0, c,
                                                   false, false);
}

constexpr int D = 768;
constexpr int S = 2048;
constexpr int B = 8;
constexpr int M = B * S;          // 16384 total rows
constexpr int NB64 = D / 64;      // 12 column blocks per projection wave row-tile

// ---------------------------------------------------------------------------
// Projection GEMM:  P[m, n] = sum_k X[m,k] * W[n,k] + bias[n]
// Per wave: 16 rows x 64 cols (4 WMMA n-tiles), K looped in 32-chunks.
// OUT_MODE 0: bf16 row-major [M, D]        (Q, K buffers)
// OUT_MODE 1: bf16 transposed [B, D, S]    (V^T buffer, packed b128 stores)
// OUT_MODE 2: f32 row-major [M, D]         (final output)
// ---------------------------------------------------------------------------
template <int OUT_MODE>
__global__ __launch_bounds__(256) void proj_kernel(const float* __restrict__ X,
                                                   const float* __restrict__ W,
                                                   const float* __restrict__ bias,
                                                   void* __restrict__ outp) {
    const int wave = (blockIdx.x * blockDim.x + threadIdx.x) >> 5;
    const int lane = threadIdx.x & 31;
    const int lo = lane & 15, hi = lane >> 4;
    const int rt = wave / NB64;          // 0..1023
    const int nb = wave % NB64;          // 0..11
    const int row0 = rt * 16;
    const int n0 = nb * 64;

    F8 acc[4];
#pragma unroll
    for (int t = 0; t < 4; ++t) acc[t].v = zero8();

    const float* xrow = X + (size_t)(row0 + lo) * D;
    for (int kc = 0; kc < D / 32; ++kc) {
        const int kb = kc * 32 + hi * 8;
        v16bf a = load_f32_frag(xrow + kb, xrow + kb + 16);
#pragma unroll
        for (int t = 0; t < 4; ++t) {
            const float* wrow = W + (size_t)(n0 + t * 16 + lo) * D;
            v16bf bm = load_f32_frag(wrow + kb, wrow + kb + 16);
            acc[t].v = wmma_bf16(a, bm, acc[t].v);
        }
    }

#pragma unroll
    for (int t = 0; t < 4; ++t) {
        const int col = n0 + t * 16 + lo;
        const float bv = bias[col];
        if (OUT_MODE == 0) {
            __bf16* out = (__bf16*)outp;
#pragma unroll
            for (int r = 0; r < 8; ++r)
                out[(size_t)(row0 + r + hi * 8) * D + col] = f2bf(acc[t].f[r] + bv);
        } else if (OUT_MODE == 1) {
            __bf16* out = (__bf16*)outp;
            const int bidx = row0 >> 11;               // row0 / S
            const int s0 = (row0 & (S - 1)) + hi * 8;  // 8 consecutive seq positions
            BF8P pk;
#pragma unroll
            for (int r = 0; r < 8; ++r) pk.e[r] = f2bf(acc[t].f[r] + bv);
            *reinterpret_cast<v8bf*>(out + (size_t)bidx * D * S + (size_t)col * S + s0) =
                pk.v;
        } else {
            float* out = (float*)outp;
#pragma unroll
            for (int r = 0; r < 8; ++r)
                out[(size_t)(row0 + r + hi * 8) * D + col] = acc[t].f[r] + bv;
        }
    }
}

// ---------------------------------------------------------------------------
// Flash attention: one wave handles (batch b, 16-query tile, 256-feature slice).
// Q,K: bf16 row-major [S,D]; V^T: bf16 [D,S]. Online softmax in exp2 domain.
// ---------------------------------------------------------------------------
__global__ __launch_bounds__(256) void attn_kernel(const __bf16* __restrict__ Q,
                                                   const __bf16* __restrict__ Kmat,
                                                   const __bf16* __restrict__ Vt,
                                                   float* __restrict__ Zsum,
                                                   int accumulate) {
    __shared__ __align__(16) __bf16 pbuf[8][16][40];  // per-wave P tile, padded stride

    const int wslot = threadIdx.x >> 5;
    const int lane = threadIdx.x & 31;
    const int lo = lane & 15, hi = lane >> 4;
    const int wave = blockIdx.x * 8 + wslot;
    const int cs = wave % 3;                 // feature slice: [cs*256, cs*256+256)
    const int qt = (wave / 3) & 127;         // query tile
    const int b = wave / (3 * 128);          // batch

    const __bf16* Qb = Q + (size_t)b * S * D;
    const __bf16* Kb = Kmat + (size_t)b * S * D;
    const __bf16* Vb = Vt + (size_t)b * D * S;
    const int row0 = qt * 16;

    F8 acc[16];
#pragma unroll
    for (int nt = 0; nt < 16; ++nt) acc[nt].v = zero8();

    float m[8], l[8];
#pragma unroll
    for (int r = 0; r < 8; ++r) { m[r] = -1.0e30f; l[r] = 0.0f; }

    // softmax computed in base-2: scores * (1/sqrt(D)) * log2(e)
    const float scl = 0.03608439182435161f * 1.4426950408889634f;

    const __bf16* qrow = Qb + (size_t)(row0 + lo) * D;

    for (int kt = 0; kt < S / 32; ++kt) {
        const int j0 = kt * 32;
        F8 s0q, s1q;
        s0q.v = zero8();
        s1q.v = zero8();
        const __bf16* krow0 = Kb + (size_t)(j0 + lo) * D;
        const __bf16* krow1 = Kb + (size_t)(j0 + 16 + lo) * D;
        for (int kc = 0; kc < D / 32; ++kc) {
            const int kb = kc * 32 + hi * 8;
            v16bf a = load_bf16_frag(qrow + kb, qrow + kb + 16);
            v16bf b0 = load_bf16_frag(krow0 + kb, krow0 + kb + 16);
            v16bf b1 = load_bf16_frag(krow1 + kb, krow1 + kb + 16);
            s0q.v = wmma_bf16(a, b0, s0q.v);
            s1q.v = wmma_bf16(a, b1, s1q.v);
        }

        // Online softmax update. Slot r on lanes [0-15]/[16-31] = query row r / r+8;
        // 16-lane xor-shuffle reductions stay within each half.
        float corr[8];
#pragma unroll
        for (int r = 0; r < 8; ++r) {
            const float x0 = s0q.f[r] * scl;
            const float x1 = s1q.f[r] * scl;
            float tm = fmaxf(x0, x1);
            tm = fmaxf(tm, __shfl_xor(tm, 1, 32));
            tm = fmaxf(tm, __shfl_xor(tm, 2, 32));
            tm = fmaxf(tm, __shfl_xor(tm, 4, 32));
            tm = fmaxf(tm, __shfl_xor(tm, 8, 32));
            const float mn = fmaxf(m[r], tm);
            corr[r] = exp2f(m[r] - mn);
            const float p0 = exp2f(x0 - mn);
            const float p1 = exp2f(x1 - mn);
            s0q.f[r] = p0;
            s1q.f[r] = p1;
            float rs = p0 + p1;
            rs += __shfl_xor(rs, 1, 32);
            rs += __shfl_xor(rs, 2, 32);
            rs += __shfl_xor(rs, 4, 32);
            rs += __shfl_xor(rs, 8, 32);
            l[r] = l[r] * corr[r] + rs;
            m[r] = mn;
        }

#pragma unroll
        for (int nt = 0; nt < 16; ++nt)
#pragma unroll
            for (int r = 0; r < 8; ++r) acc[nt].f[r] *= corr[r];

        // Re-layout P (C-layout) -> A-fragment via per-wave LDS tile.
#pragma unroll
        for (int r = 0; r < 8; ++r) {
            pbuf[wslot][r + hi * 8][lo] = f2bf(s0q.f[r]);
            pbuf[wslot][r + hi * 8][lo + 16] = f2bf(s1q.f[r]);
        }
        asm volatile("s_wait_dscnt 0x0" ::: "memory");  // same-wave LDS RAW fence
        v16bf pA =
            load_bf16_frag(&pbuf[wslot][lo][hi * 8], &pbuf[wslot][lo][hi * 8 + 16]);

        // O += P * V  : B-fragment rows are contiguous in V^T.
#pragma unroll
        for (int nt = 0; nt < 16; ++nt) {
            const __bf16* vrow = Vb + (size_t)(cs * 256 + nt * 16 + lo) * S + j0 + hi * 8;
            v16bf bv = load_bf16_frag(vrow, vrow + 16);
            acc[nt].v = wmma_bf16(pA, bv, acc[nt].v);
        }
        asm volatile("s_wait_dscnt 0x0" ::: "memory");  // WAR vs next tile's P store
    }

    float inv[8];
#pragma unroll
    for (int r = 0; r < 8; ++r) inv[r] = 1.0f / l[r];

#pragma unroll
    for (int nt = 0; nt < 16; ++nt) {
        const int col = cs * 256 + nt * 16 + lo;
#pragma unroll
        for (int r = 0; r < 8; ++r) {
            const size_t idx = ((size_t)b * S + row0 + r + hi * 8) * (size_t)D + col;
            const float val = acc[nt].f[r] * inv[r];
            if (accumulate)
                Zsum[idx] += val;
            else
                Zsum[idx] = val;
        }
    }
}

// ---------------------------------------------------------------------------
// Host-side orchestration (all on `stream`; sequential stream ordering gives
// the inter-kernel dependencies). Workspace layout (bytes):
//   [0)              qbuf  bf16 [M,D]    25,165,824
//   [+1x]            kbuf  bf16 [M,D]    25,165,824
//   [+2x]            vtbuf bf16 [B,D,S]  25,165,824
//   [+3x]            zsum  f32  [M,D]    50,331,648   -> total ~126 MB
// ---------------------------------------------------------------------------
extern "C" void kernel_launch(void* const* d_in, const int* in_sizes, int n_in,
                              void* d_out, int out_size, void* d_ws, size_t ws_size,
                              hipStream_t stream) {
    (void)in_sizes; (void)n_in; (void)out_size; (void)ws_size;

    const float* Zg = (const float*)d_in[0];
    const float* Zl = (const float*)d_in[1];
    const float* Wqg = (const float*)d_in[2];  const float* bqg = (const float*)d_in[3];
    const float* Wkl = (const float*)d_in[4];  const float* bkl = (const float*)d_in[5];
    const float* Wvl = (const float*)d_in[6];  const float* bvl = (const float*)d_in[7];
    const float* Wql = (const float*)d_in[8];  const float* bql = (const float*)d_in[9];
    const float* Wkg = (const float*)d_in[10]; const float* bkg = (const float*)d_in[11];
    const float* Wvg = (const float*)d_in[12]; const float* bvg = (const float*)d_in[13];
    const float* Wf  = (const float*)d_in[14]; const float* bf  = (const float*)d_in[15];

    const size_t bfBytes = (size_t)M * D * 2;  // bf16 buffer size
    char* ws = (char*)d_ws;
    __bf16* qb = (__bf16*)ws;
    __bf16* kb = (__bf16*)(ws + bfBytes);
    __bf16* vt = (__bf16*)(ws + 2 * bfBytes);
    float*  zs = (float*)(ws + 3 * bfBytes);

    const dim3 blk(256);
    const dim3 gproj((M / 16) * NB64 / 8);      // 1536 blocks, 8 waves each
    const dim3 gattn(B * (S / 16) * 3 / 8);     // 384 blocks

    // Direction 1: graph queries attend over lstm.
    proj_kernel<0><<<gproj, blk, 0, stream>>>(Zg, Wqg, bqg, qb);
    proj_kernel<0><<<gproj, blk, 0, stream>>>(Zl, Wkl, bkl, kb);
    proj_kernel<1><<<gproj, blk, 0, stream>>>(Zl, Wvl, bvl, vt);
    attn_kernel<<<gattn, blk, 0, stream>>>(qb, kb, vt, zs, 0);

    // Direction 2: lstm queries attend over graph (reuse buffers).
    proj_kernel<0><<<gproj, blk, 0, stream>>>(Zl, Wql, bql, qb);
    proj_kernel<0><<<gproj, blk, 0, stream>>>(Zg, Wkg, bkg, kb);
    proj_kernel<1><<<gproj, blk, 0, stream>>>(Zg, Wvg, bvg, vt);
    attn_kernel<<<gattn, blk, 0, stream>>>(qb, kb, vt, zs, 1);

    // Final fused projection -> d_out (f32).
    proj_kernel<2><<<gproj, blk, 0, stream>>>(zs, Wf, bf, (float*)d_out);
}
